// GAT0mnbn_10599979286631
// MI455X (gfx1250) — matrix operand order repaired and checked by reading, not compiled, not run on hardware
//
#include <hip/hip_runtime.h>
#include <cstdint>
#include <cstddef>

// ---------------------------------------------------------------------------
// Types for CDNA5 WMMA (gfx1250, wave32)
// ---------------------------------------------------------------------------
typedef unsigned short u16;
typedef __attribute__((ext_vector_type(16))) __bf16 v16bf;
typedef __attribute__((ext_vector_type(8)))  __bf16 v8bf;
typedef __attribute__((ext_vector_type(8)))  float  v8f;

#define SLOPE 0.2f
#define EPSBN 1e-5f

__device__ __forceinline__ u16 f32_to_bf16(float f) {
  union { float f; uint32_t u; } c; c.f = f;
  uint32_t u = c.u;
  // round-to-nearest-even
  return (u16)((u + 0x7FFFu + ((u >> 16) & 1u)) >> 16);
}

// float atomic max via signed/unsigned int atomics (init value must be -inf)
__device__ __forceinline__ void atomicMaxF(float* addr, float v) {
  if (v >= 0.0f) atomicMax((int*)addr, __float_as_int(v));
  else           atomicMin((unsigned int*)addr, __float_as_uint(v));
}

// CDNA5 async direct global->LDS copy (16B per lane), tracked by ASYNCcnt.
// lds_off: wave-relative LDS byte address (low 32 bits of a generic LDS ptr).
__device__ __forceinline__ void async_copy_b128(uint32_t lds_off, const void* gaddr) {
  asm volatile("global_load_async_to_lds_b128 %0, %1, off"
               :: "v"(lds_off), "v"(gaddr) : "memory");
}
__device__ __forceinline__ void wait_async_all() {
  asm volatile("s_wait_asynccnt 0x0" ::: "memory");
}

// ---------------------------------------------------------------------------
// f32 -> bf16 conversion
// ---------------------------------------------------------------------------
__global__ void gat_cvt_bf16(const float* __restrict__ src, u16* __restrict__ dst, long n) {
  long i = (long)blockIdx.x * blockDim.x + threadIdx.x;
  long stride = (long)gridDim.x * blockDim.x;
  for (; i < n; i += stride) dst[i] = f32_to_bf16(src[i]);
}

// ---------------------------------------------------------------------------
// WMMA bf16 GEMM:  C[M,N] = A[M,K] * B[K,N] + bias[N]
//   A, B: bf16 row-major.  C: f32 row-major.  K % 64 == 0, N % 64 == 0.
//   Block: 256 threads (8 wave32), tile 128(M) x 64(N), stage depth BK = 64.
//   Waves: 4(M) x 2(N); each wave computes 32x32 as 2x2 WMMA accumulators
//   -> 8 v_wmma per wave per LDS stage (2 K-steps x 2x2), fragments reused.
//   A tile staged via GLOBAL_LOAD_ASYNC_TO_LDS_B128 (ASYNCcnt path);
//   B tile transposed in flight via VGPR + ds_store_b16.
//   LDS: 16KB (A) + 8KB (Bt) of the WGP's 320KB.
// ---------------------------------------------------------------------------
__global__ __launch_bounds__(256) void gat_gemm_wmma_bf16(
    const u16* __restrict__ A, const u16* __restrict__ B,
    const float* __restrict__ bias, float* __restrict__ C,
    int M, int N, int K)
{
  __shared__ alignas(32) u16 As[128 * 64];  // A tile, row-major [128][64]
  __shared__ alignas(32) u16 Bt[64 * 64];   // B tile transposed [n][k]

  const int tid  = threadIdx.x;
  const int wave = tid >> 5;
  const int lane = tid & 31;
  const int m0 = blockIdx.x * 128;
  const int n0 = blockIdx.y * 64;
  const int mg = (wave & 3) * 32;           // wave M sub-tile base in As
  const int ng = (wave >> 2) * 32;          // wave N sub-tile base in Bt

  v8f acc[2][2];
  #pragma unroll
  for (int i = 0; i < 2; ++i)
    #pragma unroll
    for (int j = 0; j < 2; ++j)
      #pragma unroll
      for (int r = 0; r < 8; ++r) acc[i][j][r] = 0.0f;

  // A-staging: each thread async-copies 64B (four b128) of one row
  const int arow = tid >> 1;                // 0..127
  const int acol = (tid & 1) * 32;          // 0 or 32 (u16 units)
  int agm = m0 + arow; if (agm >= M) agm = M - 1;   // clamp: extra rows never stored
  const uint32_t aLds = (uint32_t)(uintptr_t)&As[arow * 64 + acol];
  // B-staging: each thread reads 16 consecutive N of one K row, transposes to LDS
  const int bk = tid >> 2;                  // 0..63
  const int bn = (tid & 3) * 16;            // 0..48

  for (int k0 = 0; k0 < K; k0 += 64) {
    // ---- stage A tile: global -> LDS direct (no VGPR round-trip) ----
    {
      const u16* gA = A + (size_t)agm * K + k0 + acol;
      async_copy_b128(aLds,      gA);
      async_copy_b128(aLds + 16, gA + 8);
      async_copy_b128(aLds + 32, gA + 16);
      async_copy_b128(aLds + 48, gA + 24);
    }
    // ---- stage B tile, transposed to [n][k] ----
    {
      const u16* gB = B + (size_t)(k0 + bk) * N + n0 + bn;
      u16 tmp[16];
      *(uint4*)&tmp[0] = *(const uint4*)gB;
      *(uint4*)&tmp[8] = *(const uint4*)(gB + 8);
      #pragma unroll
      for (int i = 0; i < 16; ++i) Bt[(bn + i) * 64 + bk] = tmp[i];
    }
    wait_async_all();
    __syncthreads();

    // ---- 2 K-steps x 2x2 WMMAs per stage ----
    const int am = lane & 15;
    const int kb = (lane >> 4) * 8;
    const int Kb = (lane >> 4) * 16;
    #pragma unroll
    for (int ks = 0; ks < 64; ks += 32) {
      // A fragments (contiguous k = 16*(lane/16) + s, same order as B)
      v16bf a[2];
      #pragma unroll
      for (int i = 0; i < 2; ++i) {
        const int row = mg + i * 16 + am;
        v8bf lo = *(const v8bf*)&As[row * 64 + ks + Kb];
        v8bf hi = *(const v8bf*)&As[row * 64 + ks + Kb + 8];
        #pragma unroll
        for (int t = 0; t < 8; ++t) { a[i][t] = lo[t]; a[i][8 + t] = hi[t]; }
      }
      // B fragments (lane = N%16 + 16*(K/16); K-column contiguous in Bt)
      v16bf b[2];
      #pragma unroll
      for (int j = 0; j < 2; ++j) {
        const int n = ng + j * 16 + am;
        b[j] = *(const v16bf*)&Bt[n * 64 + ks + Kb];
      }
      #pragma unroll
      for (int i = 0; i < 2; ++i)
        #pragma unroll
        for (int j = 0; j < 2; ++j)
          acc[i][j] = __builtin_amdgcn_wmma_f32_16x16x32_bf16(
              false, a[i], false, b[j], (short)0, acc[i][j], false, false);
    }
    __syncthreads();
  }

  // ---- store (32-bit C/D layout: lanes 0-15 -> M=r, lanes 16-31 -> M=8+r) ----
  const int mb = (lane >> 4) * 8;
  if (m0 + 128 <= M) {
    // full tile: unpredicated, strength-reduced row stepping
    #pragma unroll
    for (int j = 0; j < 2; ++j) {
      const int cn = n0 + ng + j * 16 + (lane & 15);
      const float bv = bias ? bias[cn] : 0.0f;
      #pragma unroll
      for (int i = 0; i < 2; ++i) {
        float* cp = C + (size_t)(m0 + mg + i * 16 + mb) * N + cn;
        #pragma unroll
        for (int r = 0; r < 8; ++r) {
          *cp = acc[i][j][r] + bv;
          cp += N;
        }
      }
    }
  } else {
    #pragma unroll
    for (int j = 0; j < 2; ++j) {
      const int cn = n0 + ng + j * 16 + (lane & 15);
      const float bv = bias ? bias[cn] : 0.0f;
      #pragma unroll
      for (int i = 0; i < 2; ++i) {
        const int rbase = m0 + mg + i * 16 + mb;
        #pragma unroll
        for (int r = 0; r < 8; ++r) {
          const int gm = rbase + r;
          if (gm < M) C[(size_t)gm * N + cn] = acc[i][j][r] + bv;
        }
      }
    }
  }
}

// ---------------------------------------------------------------------------
// init kernels
// ---------------------------------------------------------------------------
__global__ void gat_init_attn(float* __restrict__ mx, float* __restrict__ den, int n) {
  int i = blockIdx.x * blockDim.x + threadIdx.x;
  int stride = gridDim.x * blockDim.x;
  for (; i < n; i += stride) { mx[i] = -__builtin_inff(); den[i] = 0.0f; }
}

__global__ void gat_bcast_bias(float* __restrict__ out, const float* __restrict__ b,
                               long n, int mask) {
  long i = (long)blockIdx.x * blockDim.x + threadIdx.x;
  long stride = (long)gridDim.x * blockDim.x;
  for (; i < n; i += stride) out[i] = b[(int)(i & mask)];
}

// ---------------------------------------------------------------------------
// Edge pass A: logits[e,h] = sum_c leaky(xl[src]+xr[dst]+edge_attr@We) * att[h,c]
// One wave32 per (edge, head) task; We + att staged in LDS.
// Also atomic segment-max into mx[dst,h].
// ---------------------------------------------------------------------------
__global__ __launch_bounds__(256) void gat_edge_logits(
    const float* __restrict__ xl, const float* __restrict__ xr,
    const int* __restrict__ eidx, const float* __restrict__ eattr,
    const float* __restrict__ We, const float* __restrict__ att,
    float* __restrict__ logits, float* __restrict__ mx,
    int E, int Cc, int HC)
{
  extern __shared__ float smem[];          // [12*HC] We + [HC] att
  const int ED = 12;
  const int total_s = ED * HC + HC;
  for (int i = threadIdx.x; i < total_s; i += blockDim.x)
    smem[i] = (i < ED * HC) ? We[i] : att[i - ED * HC];
  __syncthreads();

  const int lane   = threadIdx.x & 31;
  const int gwave  = (blockIdx.x * blockDim.x + threadIdx.x) >> 5;
  const int nwaves = (gridDim.x * blockDim.x) >> 5;
  const int ntask  = E * 4;

  for (int t = gwave; t < ntask; t += nwaves) {
    const int e = t >> 2;
    const int h = t & 3;
    const int src = eidx[e];
    const int dst = eidx[E + e];
    const float eav = (lane < ED) ? eattr[(size_t)e * ED + lane] : 0.0f;

    const float* xls = xl + (size_t)src * HC + h * Cc;
    const float* xrs = xr + (size_t)dst * HC + h * Cc;
    const float* atts = smem + ED * HC + h * Cc;

    float part = 0.0f;
    for (int c = lane; c < Cc; c += 32) {
      float ev = 0.0f;
      #pragma unroll
      for (int k = 0; k < ED; ++k)
        ev += __shfl(eav, k, 32) * smem[k * HC + h * Cc + c];
      const float m = xls[c] + xrs[c] + ev;
      const float s = (m > 0.0f) ? m : m * SLOPE;
      part += s * atts[c];
    }
    #pragma unroll
    for (int off = 16; off > 0; off >>= 1) part += __shfl_down(part, off, 32);
    if (lane == 0) {
      logits[t] = part;
      atomicMaxF(&mx[dst * 4 + h], part);
    }
  }
}

// ---------------------------------------------------------------------------
// Edge pass B: ex = exp(logit - mx[dst]); den[dst,h] += ex
// ---------------------------------------------------------------------------
__global__ void gat_edge_exp(const float* __restrict__ logits,
                             const int* __restrict__ eidx,
                             const float* __restrict__ mx,
                             float* __restrict__ ex, float* __restrict__ den, int E)
{
  int i = blockIdx.x * blockDim.x + threadIdx.x;
  int stride = gridDim.x * blockDim.x;
  const int ntask = E * 4;
  for (; i < ntask; i += stride) {
    const int e = i >> 2;
    const int h = i & 3;
    const int dst = eidx[E + e];
    const float v = __expf(logits[i] - mx[dst * 4 + h]);
    ex[i] = v;
    atomicAdd(&den[dst * 4 + h], v);
  }
}

// ---------------------------------------------------------------------------
// Edge pass C: out[dst] += xl[src] * alpha ;  alpha = ex / (den[dst] + 1e-16)
// ---------------------------------------------------------------------------
__global__ __launch_bounds__(256) void gat_edge_scatter(
    const float* __restrict__ xl, const int* __restrict__ eidx,
    const float* __restrict__ ex, const float* __restrict__ den,
    float* __restrict__ out, int E, int Cc, int HC)
{
  const int lane   = threadIdx.x & 31;
  const int gwave  = (blockIdx.x * blockDim.x + threadIdx.x) >> 5;
  const int nwaves = (gridDim.x * blockDim.x) >> 5;
  const int ntask  = E * 4;

  for (int t = gwave; t < ntask; t += nwaves) {
    const int e = t >> 2;
    const int h = t & 3;
    const int src = eidx[e];
    const int dst = eidx[E + e];
    const float alpha = ex[t] / (den[dst * 4 + h] + 1e-16f);
    const float* xls = xl + (size_t)src * HC + h * Cc;
    float* outs = out + (size_t)dst * HC + h * Cc;
    for (int c = lane; c < Cc; c += 32)
      atomicAdd(&outs[c], xls[c] * alpha);
  }
}

// ---------------------------------------------------------------------------
// BN(relu(x)) in place; optionally emit bf16 copy for the next WMMA GEMM
// ---------------------------------------------------------------------------
__global__ void gat_relu_bn(float* __restrict__ x,
                            const float* __restrict__ g, const float* __restrict__ be,
                            const float* __restrict__ mean, const float* __restrict__ var,
                            long n, int mask, u16* __restrict__ bfout)
{
  long i = (long)blockIdx.x * blockDim.x + threadIdx.x;
  long stride = (long)gridDim.x * blockDim.x;
  for (; i < n; i += stride) {
    const int j = (int)(i & mask);
    const float r = fmaxf(x[i], 0.0f);
    const float h = (r - mean[j]) * rsqrtf(var[j] + EPSBN) * g[j] + be[j];
    x[i] = h;
    if (bfout) bfout[i] = f32_to_bf16(h);
  }
}

// ---------------------------------------------------------------------------
// Head MLP: out[g] = relu(h2[last_idx[g]] @ Wf1 + bf1) @ Wf2 + bf2
// One block of 64 threads per graph.
// ---------------------------------------------------------------------------
__global__ __launch_bounds__(64) void gat_head_mlp(
    const float* __restrict__ h2, const int* __restrict__ nn,
    const float* __restrict__ Wf1, const float* __restrict__ bf1,
    const float* __restrict__ Wf2, const float* __restrict__ bf2,
    float* __restrict__ out, int G)
{
  __shared__ float red[64];
  const int g = blockIdx.x;
  if (g >= G) return;
  int idx = 0;
  for (int k = 0; k <= g; ++k) idx += nn[k];
  idx -= 1;                                  // master node for graph g

  const int t = threadIdx.x;                 // hidden unit 0..63
  const float* hrow = h2 + (size_t)idx * 256;
  float acc = bf1[t];
  for (int k = 0; k < 256; ++k) acc += hrow[k] * Wf1[k * 64 + t];
  acc = fmaxf(acc, 0.0f) * Wf2[t];
  red[t] = acc;
  __syncthreads();
  #pragma unroll
  for (int s = 32; s > 0; s >>= 1) {
    if (t < s) red[t] += red[t + s];
    __syncthreads();
  }
  if (t == 0) out[g] = red[0] + bf2[0];
}

// ---------------------------------------------------------------------------
// Host launcher
// ---------------------------------------------------------------------------
extern "C" void kernel_launch(void* const* d_in, const int* in_sizes, int n_in,
                              void* d_out, int out_size, void* d_ws, size_t ws_size,
                              hipStream_t stream)
{
  (void)in_sizes; (void)n_in; (void)out_size; (void)ws_size;

  const int Nn = 50000, Ee = 200000, Gg = 100;
  const int INd = 256, C1c = 256, C2c = 64;
  const int HC1 = 4 * C1c;   // 1024
  const int HC2 = 4 * C2c;   // 256

  const float* x     = (const float*)d_in[0];
  const int*   eidx  = (const int*)  d_in[1];
  const float* eattr = (const float*)d_in[2];
  const int*   nn    = (const int*)  d_in[3];
  const float* Wl1 = (const float*)d_in[4];  const float* bl1 = (const float*)d_in[5];
  const float* Wr1 = (const float*)d_in[6];  const float* br1 = (const float*)d_in[7];
  const float* We1 = (const float*)d_in[8];  const float* att1= (const float*)d_in[9];
  const float* b1  = (const float*)d_in[10];
  const float* g1  = (const float*)d_in[11]; const float* be1 = (const float*)d_in[12];
  const float* m1  = (const float*)d_in[13]; const float* v1  = (const float*)d_in[14];
  const float* Wl2 = (const float*)d_in[15]; const float* bl2 = (const float*)d_in[16];
  const float* Wr2 = (const float*)d_in[17]; const float* br2 = (const float*)d_in[18];
  const float* We2 = (const float*)d_in[19]; const float* att2= (const float*)d_in[20];
  const float* b2  = (const float*)d_in[21];
  const float* g2  = (const float*)d_in[22]; const float* be2 = (const float*)d_in[23];
  const float* m2  = (const float*)d_in[24]; const float* v2  = (const float*)d_in[25];
  const float* Wf1 = (const float*)d_in[26]; const float* bf1 = (const float*)d_in[27];
  const float* Wf2 = (const float*)d_in[28]; const float* bf2 = (const float*)d_in[29];

  // ---- bump allocator over d_ws ----
  size_t off = 0;
  char* base = (char*)d_ws;
  auto alloc = [&](size_t bytes) -> void* {
    void* p = base + off;
    off = (off + bytes + 255) & ~(size_t)255;
    return p;
  };

  u16*   x_bf   = (u16*)  alloc((size_t)Nn * INd * 2);
  u16*   wl1_bf = (u16*)  alloc((size_t)INd * HC1 * 2);
  u16*   wr1_bf = (u16*)  alloc((size_t)INd * HC1 * 2);
  u16*   wl2_bf = (u16*)  alloc((size_t)HC1 * HC2 * 2);
  u16*   wr2_bf = (u16*)  alloc((size_t)HC1 * HC2 * 2);
  float* xl1    = (float*)alloc((size_t)Nn * HC1 * 4);
  float* xr1    = (float*)alloc((size_t)Nn * HC1 * 4);
  float* out1   = (float*)alloc((size_t)Nn * HC1 * 4);   // agg -> BN in place -> h1
  u16*   h1_bf  = (u16*)  alloc((size_t)Nn * HC1 * 2);
  float* xl2    = (float*)alloc((size_t)Nn * HC2 * 4);
  float* xr2    = (float*)alloc((size_t)Nn * HC2 * 4);
  float* out2   = (float*)alloc((size_t)Nn * HC2 * 4);   // agg -> BN in place -> h2
  float* logits = (float*)alloc((size_t)Ee * 4 * 4);
  float* ex     = (float*)alloc((size_t)Ee * 4 * 4);
  float* mx     = (float*)alloc((size_t)Nn * 4 * 4);
  float* den    = (float*)alloc((size_t)Nn * 4 * 4);

  const dim3 blk256(256), blk64(64);

  // ---- conversions to bf16 ----
  gat_cvt_bf16<<<2048, blk256, 0, stream>>>(x,   x_bf,   (long)Nn * INd);
  gat_cvt_bf16<<<512,  blk256, 0, stream>>>(Wl1, wl1_bf, (long)INd * HC1);
  gat_cvt_bf16<<<512,  blk256, 0, stream>>>(Wr1, wr1_bf, (long)INd * HC1);
  gat_cvt_bf16<<<512,  blk256, 0, stream>>>(Wl2, wl2_bf, (long)HC1 * HC2);
  gat_cvt_bf16<<<512,  blk256, 0, stream>>>(Wr2, wr2_bf, (long)HC1 * HC2);

  // ---- layer 1 node GEMMs (WMMA bf16, f32 accumulate) ----
  {
    dim3 grid((Nn + 127) / 128, HC1 / 64);
    gat_gemm_wmma_bf16<<<grid, blk256, 0, stream>>>(x_bf, wl1_bf, bl1, xl1, Nn, HC1, INd);
    gat_gemm_wmma_bf16<<<grid, blk256, 0, stream>>>(x_bf, wr1_bf, br1, xr1, Nn, HC1, INd);
  }

  // ---- layer 1 attention ----
  gat_init_attn<<<256, blk256, 0, stream>>>(mx, den, Nn * 4);
  gat_bcast_bias<<<2048, blk256, 0, stream>>>(out1, b1, (long)Nn * HC1, HC1 - 1);
  {
    size_t sh = (size_t)(12 * HC1 + HC1) * 4;   // 52 KB (LDS; WGP has 320 KB)
    gat_edge_logits<<<2048, blk256, sh, stream>>>(xl1, xr1, eidx, eattr, We1, att1,
                                                  logits, mx, Ee, C1c, HC1);
  }
  gat_edge_exp<<<2048, blk256, 0, stream>>>(logits, eidx, mx, ex, den, Ee);
  gat_edge_scatter<<<2048, blk256, 0, stream>>>(xl1, eidx, ex, den, out1, Ee, C1c, HC1);
  gat_relu_bn<<<4096, blk256, 0, stream>>>(out1, g1, be1, m1, v1,
                                           (long)Nn * HC1, HC1 - 1, h1_bf);

  // ---- layer 2 node GEMMs ----
  {
    dim3 grid((Nn + 127) / 128, HC2 / 64);
    gat_gemm_wmma_bf16<<<grid, blk256, 0, stream>>>(h1_bf, wl2_bf, bl2, xl2, Nn, HC2, HC1);
    gat_gemm_wmma_bf16<<<grid, blk256, 0, stream>>>(h1_bf, wr2_bf, br2, xr2, Nn, HC2, HC1);
  }

  // ---- layer 2 attention ----
  gat_init_attn<<<256, blk256, 0, stream>>>(mx, den, Nn * 4);
  gat_bcast_bias<<<2048, blk256, 0, stream>>>(out2, b2, (long)Nn * HC2, HC2 - 1);
  {
    size_t sh = (size_t)(12 * HC2 + HC2) * 4;   // 13 KB
    gat_edge_logits<<<2048, blk256, sh, stream>>>(xl2, xr2, eidx, eattr, We2, att2,
                                                  logits, mx, Ee, C2c, HC2);
  }
  gat_edge_exp<<<2048, blk256, 0, stream>>>(logits, eidx, mx, ex, den, Ee);
  gat_edge_scatter<<<2048, blk256, 0, stream>>>(xl2, eidx, ex, den, out2, Ee, C2c, HC2);
  gat_relu_bn<<<4096, blk256, 0, stream>>>(out2, g2, be2, m2, v2,
                                           (long)Nn * HC2, HC2 - 1, (u16*)nullptr);

  // ---- head MLP over master nodes ----
  gat_head_mlp<<<Gg, blk64, 0, stream>>>(out2, nn, Wf1, bf1, Wf2, bf2,
                                         (float*)d_out, Gg);
}
